// Spatial_LSTM_block_16853451670052
// MI455X (gfx1250) — matrix operand order repaired and checked by reading, not compiled, not run on hardware
//
#include <hip/hip_runtime.h>
#include <hip/hip_bf16.h>

typedef __attribute__((ext_vector_type(16))) _Float16 v16h;
typedef __attribute__((ext_vector_type(8)))  float    v8f;

#define BATCH 2048
#define CH    14
#define LEN   2048
#define HID   64
#define GATES 256          // 4 * HID
#define TTOT  960          // 512+256+128+64

// ---------------------------------------------------------------------------
// Stage 1: non-overlapping block-average pooling (kernel==stride for every
// bin). bin512 directly from X, coarser bins by halving.
// ---------------------------------------------------------------------------
__global__ __launch_bounds__(256)
void pool512_kernel(const float* __restrict__ X, float* __restrict__ P) {
    int idx = blockIdx.x * 256 + threadIdx.x;          // over BATCH*CH*512
    float4 v = ((const float4*)X)[idx];
    P[idx] = (v.x + v.y + v.z + v.w) * 0.25f;
}

__global__ __launch_bounds__(256)
void halve_kernel(const float* __restrict__ src, float* __restrict__ dst) {
    int idx = blockIdx.x * 256 + threadIdx.x;
    float2 v = ((const float2*)src)[idx];
    dst[idx] = 0.5f * (v.x + v.y);
}

// ---------------------------------------------------------------------------
// Stage 2: conv1d(14->4, k=3, pad=1) + LeakyReLU, time-major output
// xs[tg*BATCH*4 + b*4 + f]. Each 256-thread block sits in one tg -> one bin.
// ---------------------------------------------------------------------------
__global__ __launch_bounds__(256)
void conv_kernel(const float* __restrict__ p512, const float* __restrict__ p256,
                 const float* __restrict__ p128, const float* __restrict__ p64,
                 const float* __restrict__ cw,   const float* __restrict__ cb,
                 float* __restrict__ xs) {
    __shared__ float wsm[4 * CH * 3];
    __shared__ float bsm[4];
    int idx = blockIdx.x * 256 + threadIdx.x;
    int tg  = idx >> 11;                 // uniform across the block
    int b   = idx & 2047;
    int bin, t, T; const float* P;
    if (tg < 512)      { bin = 0; t = tg;       T = 512; P = p512; }
    else if (tg < 768) { bin = 1; t = tg - 512; T = 256; P = p256; }
    else if (tg < 896) { bin = 2; t = tg - 768; T = 128; P = p128; }
    else               { bin = 3; t = tg - 896; T = 64;  P = p64;  }
    if (threadIdx.x < 4 * CH * 3) wsm[threadIdx.x] = cw[bin * 4 * CH * 3 + threadIdx.x];
    if (threadIdx.x < 4)          bsm[threadIdx.x] = cb[bin * 4 + threadIdx.x];
    __syncthreads();

    float a0 = bsm[0], a1 = bsm[1], a2 = bsm[2], a3 = bsm[3];
    for (int c = 0; c < CH; ++c) {
        const float* row = P + ((size_t)b * CH + c) * T + t;
        float xm = (t > 0)     ? row[-1] : 0.f;
        float x0 = row[0];
        float xp = (t < T - 1) ? row[1]  : 0.f;
        a0 += xm * wsm[(0*CH+c)*3+0] + x0 * wsm[(0*CH+c)*3+1] + xp * wsm[(0*CH+c)*3+2];
        a1 += xm * wsm[(1*CH+c)*3+0] + x0 * wsm[(1*CH+c)*3+1] + xp * wsm[(1*CH+c)*3+2];
        a2 += xm * wsm[(2*CH+c)*3+0] + x0 * wsm[(2*CH+c)*3+1] + xp * wsm[(2*CH+c)*3+2];
        a3 += xm * wsm[(3*CH+c)*3+0] + x0 * wsm[(3*CH+c)*3+1] + xp * wsm[(3*CH+c)*3+2];
    }
    float4 r;
    r.x = (a0 >= 0.f) ? a0 : 0.01f * a0;
    r.y = (a1 >= 0.f) ? a1 : 0.01f * a1;
    r.z = (a2 >= 0.f) ? a2 : 0.01f * a2;
    r.w = (a3 >= 0.f) ? a3 : 0.01f * a3;
    ((float4*)xs)[idx] = r;
}

// ---------------------------------------------------------------------------
// Stage 3: LSTM. One WG = one 16-row batch tile of one bin, persistent over
// all T steps. 16 waves; wave w owns gate columns [16w, 16w+16). Per step:
//   gates = [h | x] @ [W_hh ; W_ih]^T  via 3x v_wmma_f32_16x16x32_f16
// (K=64 hidden in 2 WMMAs + K=4 input folded into a 3rd mostly-zero WMMA).
// W fragments are built in registers ONCE (loop-invariant); only the A (h)
// fragments and the 16x4 x-tile touch LDS inside the recurrence.
// ---------------------------------------------------------------------------
__global__ __launch_bounds__(512, 1)
void lstm_kernel(const float* __restrict__ xs,
                 const float* __restrict__ whh_g,   // (4,256,64)
                 const float* __restrict__ wih_g,   // (4,256,4)
                 const float* __restrict__ bih_g,   // (4,256)
                 const float* __restrict__ bhh_g,   // (4,256)
                 const float* __restrict__ lw_g,    // (4,1,64)
                 const float* __restrict__ lb_g,    // (4,1)
                 float* __restrict__ out) {
    __shared__ float    act[GATES * 16];    // 16 KB  activated gates [n][m]
    __shared__ _Float16 hh[16 * HID];       //  2 KB  hidden state [m][k] f16
    __shared__ float    cc[16 * HID];       //  4 KB  cell state   [m][k] f32
    __shared__ _Float16 xt[16 * 4];         //  128 B conv input tile (f16)
    __shared__ float    lw[HID];

    const int tid  = threadIdx.x;
    const int bin  = blockIdx.y;
    const int b0   = blockIdx.x * 16;
    const int T    = 512 >> bin;
    const int toff = (bin == 0) ? 0 : (1024 - (1024 >> bin)); // 0,512,768,896

    if (tid < HID) lw[tid] = lw_g[bin * HID + tid];
    for (int i = tid; i < 16 * HID; i += 512) { hh[i] = (_Float16)0.f; cc[i] = 0.f; }

    const int  wave = tid >> 5;
    const int  lane = tid & 31;
    const int  mrow = lane & 15;
    const bool lo   = lane < 16;
    const int  kbA  = lo ? 0 : 16;    // A 16x32 f16: K = kbA..kbA+15 (contiguous, matches B)
    const int  kbB  = lo ? 0 : 16;    // B 32x16 f16 layout
    const int  moff = lo ? 0 : 8;     // C/D layout: M = j + 8*(lane>=16)
    const int  ncol = wave * 16 + (lane & 15);   // gate column 0..255
    const int  gate = wave >> 2;                 // 0=i 1=f 2=g 3=o

    // Branch-free activation: sigmoid(x) = rcp(1+exp(-x)); tanh = 2*sig(2x)-1.
    const float asc = (gate == 2) ? 2.f : 1.f;   // argument scale
    const float msc = (gate == 2) ? 2.f : 1.f;   // result scale
    const float off = (gate == 2) ? -1.f : 0.f;  // result offset
    const float bb  = bih_g[bin * GATES + ncol] + bhh_g[bin * GATES + ncol];

    // ---- loop-invariant weight fragments (registers, from global) ----
    union frag { v16h v; _Float16 h[16]; };
    frag B0, B1, B2;
    {
        const float* wrow = whh_g + (size_t)bin * GATES * HID + (size_t)ncol * HID;
#pragma unroll
        for (int j = 0; j < 16; ++j) {
            B0.h[j] = (_Float16)wrow[kbB + j];        // K = kbB..kbB+15
            B1.h[j] = (_Float16)wrow[32 + kbB + j];   // K = 32+kbB..
            B2.h[j] = (_Float16)0.f;
        }
        if (lo) {                                     // W_ih lives in K=0..3
            const float* xrow = wih_g + (size_t)bin * GATES * 4 + (size_t)ncol * 4;
#pragma unroll
            for (int j = 0; j < 4; ++j) B2.h[j] = (_Float16)xrow[j];
        }
    }
    frag A2;                                          // x-tile fragment
#pragma unroll
    for (int j = 0; j < 16; ++j) A2.h[j] = (_Float16)0.f;

    __syncthreads();

    for (int t = 0; t < T; ++t) {
        if (tid < 64) {
            size_t base = (size_t)(toff + t) * (BATCH * 4) + (size_t)b0 * 4;
            xt[tid] = (_Float16)xs[base + tid];
            if (t + 1 < T)
                __builtin_prefetch(&xs[base + (size_t)(BATCH * 4) + tid], 0, 0);
        }
        __syncthreads();

        frag A0, A1;
#pragma unroll
        for (int j = 0; j < 8; ++j) {
            A0.h[j]     = hh[mrow * HID + kbA + j];
            A0.h[8 + j] = hh[mrow * HID + 8 + kbA + j];
            A1.h[j]     = hh[mrow * HID + 32 + kbA + j];
            A1.h[8 + j] = hh[mrow * HID + 40 + kbA + j];
        }
        if (lo) {                                     // x in K=0..3, lanes 0-15
#pragma unroll
            for (int j = 0; j < 4; ++j) A2.h[j] = xt[mrow * 4 + j];
        }

        v8f acc = {};
        acc = __builtin_amdgcn_wmma_f32_16x16x32_f16(false, A0.v, false, B0.v,
                                                     (short)0, acc, false, false);
        acc = __builtin_amdgcn_wmma_f32_16x16x32_f16(false, A1.v, false, B1.v,
                                                     (short)0, acc, false, false);
        acc = __builtin_amdgcn_wmma_f32_16x16x32_f16(false, A2.v, false, B2.v,
                                                     (short)0, acc, false, false);
        union { v8f v; float f[8]; } C; C.v = acc;

#pragma unroll
        for (int j = 0; j < 8; ++j) {
            float g = C.f[j] + bb;
            float s = __builtin_amdgcn_rcpf(1.f + __expf(-asc * g));
            act[ncol * 16 + (moff + j)] = __builtin_fmaf(msc, s, off);
        }
        __syncthreads();

        // --- c,h update (f32 state), repack h to f16 for next matmul ---
        for (int idx = tid; idx < 16 * HID; idx += 512) {
            int m = idx & 15, k = idx >> 4;
            float iv = act[(0   + k) * 16 + m];
            float fv = act[(64  + k) * 16 + m];
            float gv = act[(128 + k) * 16 + m];
            float ov = act[(192 + k) * 16 + m];
            float c  = fv * cc[m * HID + k] + iv * gv;
            cc[m * HID + k] = c;
            // tanh(c) = 2*sigmoid(2c)-1, branch-free
            float th = __builtin_fmaf(2.f,
                        __builtin_amdgcn_rcpf(1.f + __expf(-2.f * c)), -1.f);
            hh[m * HID + k] = (_Float16)(ov * th);
        }
        __syncthreads();
    }

    // final linear head: out[b, bin]
    if (tid < 16) {
        float s = lb_g[bin];
        for (int k = 0; k < HID; ++k) s += (float)hh[tid * HID + k] * lw[k];
        out[(b0 + tid) * 4 + bin] = s;
    }
}

// ---------------------------------------------------------------------------
extern "C" void kernel_launch(void* const* d_in, const int* in_sizes, int n_in,
                              void* d_out, int out_size, void* d_ws, size_t ws_size,
                              hipStream_t stream) {
    const float* X      = (const float*)d_in[0];
    const float* conv_w = (const float*)d_in[1];
    const float* conv_b = (const float*)d_in[2];
    const float* w_ih   = (const float*)d_in[3];
    const float* w_hh   = (const float*)d_in[4];
    const float* b_ih   = (const float*)d_in[5];
    const float* b_hh   = (const float*)d_in[6];
    const float* lin_w  = (const float*)d_in[7];
    const float* lin_b  = (const float*)d_in[8];
    float* out = (float*)d_out;

    float* ws   = (float*)d_ws;
    float* p512 = ws;                                   // 2048*14*512
    float* p256 = p512 + (size_t)BATCH * CH * 512;
    float* p128 = p256 + (size_t)BATCH * CH * 256;
    float* p64  = p128 + (size_t)BATCH * CH * 128;
    float* xs   = p64  + (size_t)BATCH * CH * 64;       // 960*2048*4 time-major

    pool512_kernel<<<(BATCH * CH * 512) / 256, 256, 0, stream>>>(X, p512);
    halve_kernel  <<<(BATCH * CH * 256) / 256, 256, 0, stream>>>(p512, p256);
    halve_kernel  <<<(BATCH * CH * 128) / 256, 256, 0, stream>>>(p256, p128);
    halve_kernel  <<<(BATCH * CH *  64) / 256, 256, 0, stream>>>(p128, p64);

    conv_kernel<<<(TTOT * BATCH) / 256, 256, 0, stream>>>(p512, p256, p128, p64,
                                                          conv_w, conv_b, xs);

    dim3 grid(BATCH / 16, 4);
    lstm_kernel<<<grid, 512, 0, stream>>>(xs, w_hh, w_ih, b_ih, b_hh,
                                          lin_w, lin_b, out);
}
